// AttnDecoderRNN_19215683682782
// MI455X (gfx1250) — compile-verified, matrix operand order checked by reading, not run on hardware
//
#include <hip/hip_runtime.h>

// Problem constants (from the reference)
#define K_Y   30000
#define T_Y   64
#define M_DIM 620
#define N_DIM 1024
#define P_DIM 1024
#define L_DIM 512
#define T_X   80
#define SOS   1
#define N2    2048   // 2*N
#define L2DIM 1024   // 2*L

typedef __attribute__((ext_vector_type(16))) _Float16     v16h;
typedef __attribute__((ext_vector_type(8)))  float        v8f;
typedef __attribute__((ext_vector_type(4)))  unsigned int u32x4;
typedef __attribute__((ext_vector_type(8)))  int          i32x8;
typedef __attribute__((ext_vector_type(4)))  int          i32x4;

// ---------------------------------------------------------------------------
// One-time: s0 = tanh(h_B_Tx @ Ws + bWs); y0 = SOS
// ---------------------------------------------------------------------------
__global__ void k_init(const float* __restrict__ hB,
                       const float* __restrict__ Ws,
                       const float* __restrict__ bWs,
                       float* __restrict__ s, int* __restrict__ yDev) {
  __shared__ float hsh[N_DIM];
  int tid = threadIdx.x;
  for (int i = tid; i < N_DIM; i += blockDim.x) hsh[i] = hB[i];
  __syncthreads();
  int j = blockIdx.x * blockDim.x + tid;
  float a = bWs[j];
  for (int k = 0; k < N_DIM; ++k) a += hsh[k] * Ws[k * N_DIM + j];
  s[j] = tanhf(a);
  if (j == 0) yDev[0] = SOS;
}

// ---------------------------------------------------------------------------
// One-time GEMM via WMMA + TDM: Uah[80,1024] = h[80,2048] @ Ua[2048,1024]+bUa
// Grid: 80 blocks x 128 threads (4 wave32).  Block -> one 16-row M tile and a
// 64-column N panel (4 waves x 16 cols).  Per K-tile of 64:
//   wave0 issues TENSOR_LOAD_TO_LDS of Ua[kk:kk+64, col0:col0+64] (16 KB),
//   s_wait_tensorcnt 0, barrier, then each wave runs 2 x v_wmma_f32_16x16x32_f16.
// ---------------------------------------------------------------------------
#define KTILE 64
#define BPANEL 64   // columns per block

__global__ void k_uah_wmma(const float* __restrict__ h,
                           const float* __restrict__ Ua,
                           const float* __restrict__ bUa,
                           float* __restrict__ Uah) {
  __shared__ float Bsh[KTILE * BPANEL];   // 16 KB

  int waveInBlk = threadIdx.x >> 5;                 // 0..3
  int lane      = threadIdx.x & 31;
  int laneHalf  = lane >> 4;                        // 0: lanes 0-15, 1: 16-31
  int l15       = lane & 15;

  int mtile = blockIdx.x >> 4;                      // 0..4  (80 = 5*16 rows)
  int ntile = (blockIdx.x & 15) * 4 + waveInBlk;    // 0..63
  int col0  = (blockIdx.x & 15) * BPANEL;           // first column of panel

  int mrow = mtile * 16 + l15;                      // A row held by this lane
  int ncol = ntile * 16 + l15;                      // B/C column held by lane
  int lcol = ncol - col0;                           // column within LDS panel

  v8f acc;
  float bias = bUa[ncol];
#pragma unroll
  for (int g = 0; g < 8; ++g) acc[g] = bias;

  const float* hrow = h + mrow * N2;
  unsigned int ldsAddr = (unsigned int)(unsigned long long)(uintptr_t)&Bsh[0];

  for (int kk = 0; kk < N2; kk += KTILE) {
    // ---- stage B panel Ua[kk:kk+64, col0:col0+64] into LDS ----
#if defined(__has_builtin) && __has_builtin(__builtin_amdgcn_tensor_load_to_lds)
    if (waveInBlk == 0) {
      unsigned long long gaddr =
          (unsigned long long)(uintptr_t)(Ua + (long)kk * P_DIM + col0);
      u32x4 g0;
      g0[0] = 1u;                                   // count=1, user descriptor
      g0[1] = ldsAddr;                              // lds_addr (bytes)
      g0[2] = (unsigned int)(gaddr & 0xFFFFFFFFull);        // global_addr lo
      g0[3] = (unsigned int)((gaddr >> 32) & 0x01FFFFFFull) // global_addr hi
              | (2u << 30);                         // type = 2 (image)
      i32x8 g1;
      g1[0] = (int)(2u << 16);                      // data_size=4B, no mask
      g1[1] = (int)((P_DIM & 0xFFFF) << 16);        // tensor_dim0 = 1024 (lo16)
      g1[2] = (int)((N2 & 0xFFFF) << 16);           // tensor_dim1 = 2048 (lo16)
      g1[3] = (int)(BPANEL << 16);                  // tile_dim0 = 64
      g1[4] = KTILE;                                // tile_dim1 = 64
      g1[5] = P_DIM;                                // tensor_dim0_stride = 1024
      g1[6] = 0;
      g1[7] = 0;
      i32x4 gz = {0, 0, 0, 0};
#if defined(__clang_major__) && (__clang_major__ >= 23)
      i32x8 gz8 = {0, 0, 0, 0, 0, 0, 0, 0};
      __builtin_amdgcn_tensor_load_to_lds(g0, g1, gz, gz, gz8, 0);
#else
      __builtin_amdgcn_tensor_load_to_lds(g0, g1, gz, gz, 0);
#endif
      __builtin_amdgcn_s_wait_tensorcnt(0);
    }
#else
    // Fallback: cooperative global->LDS fill
    for (int i = threadIdx.x; i < KTILE * BPANEL; i += blockDim.x)
      Bsh[i] = Ua[(long)(kk + (i / BPANEL)) * P_DIM + col0 + (i % BPANEL)];
#endif
    __syncthreads();

    // ---- two 16x16x32 WMMAs over this K-tile ----
#pragma unroll
    for (int ks = 0; ks < KTILE; ks += 32) {
      v16h a, b;
      int kA = kk + ks + laneHalf * 8;   // global K base for this lane (A)
      int kL = ks + laneHalf * 8;        // LDS K base for this lane (B)
#pragma unroll
      for (int g = 0; g < 4; ++g) {
        a[2 * g]         = (_Float16)hrow[kA + 2 * g];
        a[2 * g + 1]     = (_Float16)hrow[kA + 2 * g + 1];
        a[8 + 2 * g]     = (_Float16)hrow[kA + 16 + 2 * g];
        a[8 + 2 * g + 1] = (_Float16)hrow[kA + 16 + 2 * g + 1];
        b[2 * g]         = (_Float16)Bsh[(kL + 2 * g) * BPANEL + lcol];
        b[2 * g + 1]     = (_Float16)Bsh[(kL + 2 * g + 1) * BPANEL + lcol];
        b[8 + 2 * g]     = (_Float16)Bsh[(kL + 16 + 2 * g) * BPANEL + lcol];
        b[8 + 2 * g + 1] = (_Float16)Bsh[(kL + 16 + 2 * g + 1) * BPANEL + lcol];
      }
      acc = __builtin_amdgcn_wmma_f32_16x16x32_f16(
          /*neg_a=*/false, a, /*neg_b=*/false, b,
          /*c_mod=*/(short)0, acc, /*reuse_a=*/false, /*reuse_b=*/false);
    }
    __syncthreads();   // protect Bsh before next TDM overwrite
  }

#pragma unroll
  for (int g = 0; g < 8; ++g) {
    int row = mtile * 16 + laneHalf * 8 + g;  // C layout: VGPR g -> M=g / 8+g
    Uah[row * P_DIM + ncol] = acc[g];
  }
}

// ---------------------------------------------------------------------------
// Per step: attention. Was = s@Wa+bWa; e[t] = tanh(Was+Uah[t]) . va + bva;
// alpha = softmax(e); c = sum_t alpha[t] * h[t].  One 1024-thread workgroup.
// ---------------------------------------------------------------------------
__global__ void k_attn(const float* __restrict__ s,
                       const float* __restrict__ Wa,
                       const float* __restrict__ bWa,
                       const float* __restrict__ Uah,
                       const float* __restrict__ va,
                       const float* __restrict__ bva,
                       const float* __restrict__ h,
                       float* __restrict__ c_out) {
  __shared__ float s_sh[N_DIM];
  __shared__ float was_sh[P_DIM];
  __shared__ float e_sh[T_X];
  int tid = threadIdx.x;
  s_sh[tid] = s[tid];
  __syncthreads();

  float a = bWa[tid];
  for (int k = 0; k < N_DIM; ++k) a += s_sh[k] * Wa[k * P_DIM + tid];
  was_sh[tid] = a;
  __syncthreads();

  int wave = tid >> 5, lane = tid & 31;
  for (int t = wave; t < T_X; t += 32) {
    float p = 0.f;
    const float* ur = Uah + t * P_DIM;
    for (int k = lane; k < P_DIM; k += 32)
      p += tanhf(was_sh[k] + ur[k]) * va[k];
    for (int off = 16; off; off >>= 1) p += __shfl_xor(p, off, 32);
    if (lane == 0) e_sh[t] = p + bva[0];
  }
  __syncthreads();

  if (wave == 0) {  // softmax over 80 in one wave32
    float m = -1e30f;
    for (int t = lane; t < T_X; t += 32) m = fmaxf(m, e_sh[t]);
    for (int off = 16; off; off >>= 1) m = fmaxf(m, __shfl_xor(m, off, 32));
    float ssum = 0.f;
    for (int t = lane; t < T_X; t += 32) ssum += expf(e_sh[t] - m);
    for (int off = 16; off; off >>= 1) ssum += __shfl_xor(ssum, off, 32);
    float inv = 1.f / ssum;
    for (int t = lane; t < T_X; t += 32) e_sh[t] = expf(e_sh[t] - m) * inv;
  }
  __syncthreads();

  for (int d = tid; d < N2; d += 1024) {
    float cc = 0.f;
    for (int t = 0; t < T_X; ++t) cc += e_sh[t] * h[t * N2 + d];
    c_out[d] = cc;
  }
}

// ---------------------------------------------------------------------------
// Per step: r, z gates + deep-output preact t2 (all use s_{i-1}).
// 12 blocks x 256: blocks 0-3 -> r, 4-7 -> z, 8-11 -> t2.
// ---------------------------------------------------------------------------
__global__ void k_gates(const float* __restrict__ emb, const int* __restrict__ yDev,
                        const float* __restrict__ s, const float* __restrict__ c,
                        const float* __restrict__ Wr, const float* __restrict__ bWr,
                        const float* __restrict__ Ur, const float* __restrict__ bUr,
                        const float* __restrict__ Cr, const float* __restrict__ bCr,
                        const float* __restrict__ Wz, const float* __restrict__ bWz,
                        const float* __restrict__ Uz, const float* __restrict__ bUz,
                        const float* __restrict__ Cz, const float* __restrict__ bCz,
                        const float* __restrict__ Vo, const float* __restrict__ bVo,
                        const float* __restrict__ Uo, const float* __restrict__ bUo,
                        const float* __restrict__ Co, const float* __restrict__ bCo,
                        float* __restrict__ r, float* __restrict__ zg,
                        float* __restrict__ t2) {
  __shared__ float sh[M_DIM + N_DIM + N2];
  float* eysh = sh;
  float* ssh  = sh + M_DIM;
  float* csh  = ssh + N_DIM;
  int tid = threadIdx.x;
  const float* ey = emb + (long)yDev[0] * M_DIM;
  for (int i = tid; i < M_DIM; i += blockDim.x) eysh[i] = ey[i];
  for (int i = tid; i < N_DIM; i += blockDim.x) ssh[i]  = s[i];
  for (int i = tid; i < N2;    i += blockDim.x) csh[i]  = c[i];
  __syncthreads();

  int sel = blockIdx.x >> 2;
  int j   = (blockIdx.x & 3) * blockDim.x + tid;  // 0..1023
  const float *W, *U, *C, *bW, *bU, *bC;
  if (sel == 0)      { W = Wr; U = Ur; C = Cr; bW = bWr; bU = bUr; bC = bCr; }
  else if (sel == 1) { W = Wz; U = Uz; C = Cz; bW = bWz; bU = bUz; bC = bCz; }
  else               { W = Vo; U = Uo; C = Co; bW = bVo; bU = bUo; bC = bCo; }

  float a = bW[j] + bU[j] + bC[j];
  for (int k = 0; k < M_DIM; ++k) a += eysh[k] * W[k * N_DIM + j];
  for (int k = 0; k < N_DIM; ++k) a += ssh[k]  * U[k * N_DIM + j];
  for (int k = 0; k < N2;    ++k) a += csh[k]  * C[k * N_DIM + j];

  if (sel == 0)      r[j]  = 1.f / (1.f + expf(-a));
  else if (sel == 1) zg[j] = 1.f / (1.f + expf(-a));
  else               t2[j] = a;
}

// ---------------------------------------------------------------------------
// Per step: n = tanh(Ey@Wn + (r*s)@Un + c@Cn + b); snew = (1-z)*s + z*n;
// maxout tmx[i] = max(t2[2i], t2[2i+1]).  4 blocks x 256.
// ---------------------------------------------------------------------------
__global__ void k_state(const float* __restrict__ emb, const int* __restrict__ yDev,
                        const float* __restrict__ s, const float* __restrict__ c,
                        const float* __restrict__ r, const float* __restrict__ zg,
                        const float* __restrict__ Wn, const float* __restrict__ bWn,
                        const float* __restrict__ Un, const float* __restrict__ bUn,
                        const float* __restrict__ Cn, const float* __restrict__ bCn,
                        const float* __restrict__ t2,
                        float* __restrict__ snew, float* __restrict__ tmx) {
  __shared__ float sh[M_DIM + N_DIM + N2];
  float* eysh = sh;
  float* rssh = sh + M_DIM;
  float* csh  = rssh + N_DIM;
  int tid = threadIdx.x;
  const float* ey = emb + (long)yDev[0] * M_DIM;
  for (int i = tid; i < M_DIM; i += blockDim.x) eysh[i] = ey[i];
  for (int i = tid; i < N_DIM; i += blockDim.x) rssh[i] = r[i] * s[i];
  for (int i = tid; i < N2;    i += blockDim.x) csh[i]  = c[i];
  __syncthreads();

  int j = blockIdx.x * blockDim.x + tid;
  float a = bWn[j] + bUn[j] + bCn[j];
  for (int k = 0; k < M_DIM; ++k) a += eysh[k] * Wn[k * N_DIM + j];
  for (int k = 0; k < N_DIM; ++k) a += rssh[k] * Un[k * N_DIM + j];
  for (int k = 0; k < N2;    ++k) a += csh[k]  * Cn[k * N_DIM + j];
  float n = tanhf(a);
  float z = zg[j];
  snew[j] = (1.f - z) * s[j] + z * n;
  if (j < L_DIM) tmx[j] = fmaxf(t2[2 * j], t2[2 * j + 1]);
}

// ---------------------------------------------------------------------------
// Per step: Py = tmx @ Wo + bWo  (GEMV, 30000 outputs over K=512)
// ---------------------------------------------------------------------------
__global__ void k_py(const float* __restrict__ tmx, const float* __restrict__ Wo,
                     const float* __restrict__ bWo, float* __restrict__ Py) {
  __shared__ float tsh[L_DIM];
  int tid = threadIdx.x;
  for (int i = tid; i < L_DIM; i += blockDim.x) tsh[i] = tmx[i];
  __syncthreads();
  int j = blockIdx.x * blockDim.x + tid;
  if (j >= K_Y) return;
  float a = bWo[j];
  for (int k = 0; k < L_DIM; ++k) {
    if ((k & 63) == 0 && (k + 64) < L_DIM)
      __builtin_prefetch(&Wo[(k + 64) * K_Y + j], 0, 0);  // global_prefetch_b8
    a += tsh[k] * Wo[k * K_Y + j];
  }
  Py[j] = a;
}

// ---------------------------------------------------------------------------
// Per step: reduce Py -> (max, log-sum-exp, argmax); commit s <- snew,
// y <- argmax. Single 1024-thread workgroup.
// ---------------------------------------------------------------------------
__global__ void k_red(const float* __restrict__ Py, float* __restrict__ red,
                      int* __restrict__ yDev, float* __restrict__ s,
                      const float* __restrict__ snew) {
  __shared__ float mred[1024];
  __shared__ int   ired[1024];
  int tid = threadIdx.x;
  s[tid] = snew[tid];  // commit next-state (all step consumers already ran)

  float m = -1e30f; int mi = 0;
  for (int j = tid; j < K_Y; j += 1024) {
    float v = Py[j];
    if (v > m) { m = v; mi = j; }
  }
  mred[tid] = m; ired[tid] = mi;
  __syncthreads();
  for (int off = 512; off; off >>= 1) {
    if (tid < off) {
      float vo = mred[tid + off];
      if (vo > mred[tid] || (vo == mred[tid] && ired[tid + off] < ired[tid])) {
        mred[tid] = vo; ired[tid] = ired[tid + off];
      }
    }
    __syncthreads();
  }
  float gmax = mred[0];
  int   garg = ired[0];
  __syncthreads();

  float ssum = 0.f;
  for (int j = tid; j < K_Y; j += 1024) ssum += expf(Py[j] - gmax);
  mred[tid] = ssum;
  __syncthreads();
  for (int off = 512; off; off >>= 1) {
    if (tid < off) mred[tid] += mred[tid + off];
    __syncthreads();
  }
  if (tid == 0) {
    red[0] = gmax;
    red[1] = logf(mred[0]);
    yDev[0] = garg;
  }
}

// ---------------------------------------------------------------------------
// Per step: out = Py - max - logsumexp
// ---------------------------------------------------------------------------
__global__ void k_write(const float* __restrict__ Py, const float* __restrict__ red,
                        float* __restrict__ out) {
  int j = blockIdx.x * blockDim.x + threadIdx.x;
  if (j < K_Y) out[j] = Py[j] - red[0] - red[1];
}

// ---------------------------------------------------------------------------
extern "C" void kernel_launch(void* const* d_in, const int* in_sizes, int n_in,
                              void* d_out, int out_size, void* d_ws, size_t ws_size,
                              hipStream_t stream) {
  const float* h   = (const float*)d_in[0];
  const float* hB  = (const float*)d_in[1];
  const float* emb = (const float*)d_in[2];
  const float* Ws  = (const float*)d_in[3];
  const float* bWs = (const float*)d_in[4];
  const float* Wa  = (const float*)d_in[5];
  const float* bWa = (const float*)d_in[6];
  const float* Ua  = (const float*)d_in[7];
  const float* bUa = (const float*)d_in[8];
  const float* va  = (const float*)d_in[9];
  const float* bva = (const float*)d_in[10];
  const float* Wr  = (const float*)d_in[11];
  const float* bWr = (const float*)d_in[12];
  const float* Ur  = (const float*)d_in[13];
  const float* bUr = (const float*)d_in[14];
  const float* Cr  = (const float*)d_in[15];
  const float* bCr = (const float*)d_in[16];
  const float* Wz  = (const float*)d_in[17];
  const float* bWz = (const float*)d_in[18];
  const float* Uz  = (const float*)d_in[19];
  const float* bUz = (const float*)d_in[20];
  const float* Cz  = (const float*)d_in[21];
  const float* bCz = (const float*)d_in[22];
  const float* Wn  = (const float*)d_in[23];
  const float* bWn = (const float*)d_in[24];
  const float* Un  = (const float*)d_in[25];
  const float* bUn = (const float*)d_in[26];
  const float* Cn  = (const float*)d_in[27];
  const float* bCn = (const float*)d_in[28];
  const float* Wo  = (const float*)d_in[29];
  const float* bWo = (const float*)d_in[30];
  const float* Vo  = (const float*)d_in[31];
  const float* bVo = (const float*)d_in[32];
  const float* Uo  = (const float*)d_in[33];
  const float* bUo = (const float*)d_in[34];
  const float* Co  = (const float*)d_in[35];
  const float* bCo = (const float*)d_in[36];

  float* out = (float*)d_out;
  float* ws  = (float*)d_ws;
  float* Uah  = ws;  ws += T_X * P_DIM;   // 81920
  float* s    = ws;  ws += N_DIM;
  float* snew = ws;  ws += N_DIM;
  float* c    = ws;  ws += N2;
  float* r    = ws;  ws += N_DIM;
  float* zg   = ws;  ws += N_DIM;
  float* t2   = ws;  ws += L2DIM;
  float* tmx  = ws;  ws += L_DIM;
  float* Py   = ws;  ws += K_Y;
  float* red  = ws;  ws += 2;
  int*   yDev = (int*)ws;

  k_init<<<N_DIM / 256, 256, 0, stream>>>(hB, Ws, bWs, s, yDev);
  k_uah_wmma<<<80, 128, 0, stream>>>(h, Ua, bUa, Uah);

  for (int step = 0; step < T_Y; ++step) {
    k_attn<<<1, 1024, 0, stream>>>(s, Wa, bWa, Uah, va, bva, h, c);
    k_gates<<<12, 256, 0, stream>>>(emb, yDev, s, c,
        Wr, bWr, Ur, bUr, Cr, bCr,
        Wz, bWz, Uz, bUz, Cz, bCz,
        Vo, bVo, Uo, bUo, Co, bCo,
        r, zg, t2);
    k_state<<<4, 256, 0, stream>>>(emb, yDev, s, c, r, zg,
        Wn, bWn, Un, bUn, Cn, bCn, t2, snew, tmx);
    k_py<<<(K_Y + 255) / 256, 256, 0, stream>>>(tmx, Wo, bWo, Py);
    k_red<<<1, 1024, 0, stream>>>(Py, red, yDev, s, snew);
    k_write<<<(K_Y + 255) / 256, 256, 0, stream>>>(Py, red, out + (long)step * K_Y);
  }
}
